// CPGMD_13082470384032
// MI455X (gfx1250) — matrix operand order repaired and checked by reading, not compiled, not run on hardware
//
#include <hip/hip_runtime.h>
#include <math.h>

// ---------------- model dims ----------------
#define BB   16
#define CIN  4
#define IMG  32
#define PP   2
#define DD   768
#define DEPTH 8
#define NN   16
#define SS   256
#define HEADS 8
#define HD   96          // DD / HEADS
#define DI   1536        // 2*DD
#define DS2  16
#define DTR  48
#define DCONV 4
#define OUTC 8           // 2*CIN
#define FREQ 256
#define TOK  (BB*SS)     // 4096

typedef _Float16 v16h __attribute__((ext_vector_type(16)));
typedef _Float16 v8h  __attribute__((ext_vector_type(8)));
typedef float    v8f  __attribute__((ext_vector_type(8)));

__device__ __forceinline__ float siluf(float v) { return v / (1.f + __expf(-v)); }

// =====================================================================
// Generic batched NT GEMM:  C[M,N] = act( A[M,K] * Wt[N,K]^T + bias[N] )
// A, Wt f16 row-major. Each wave computes a (TM*16) x (TN*16) macro-tile,
// K step 32, register-blocked: TM*TN WMMAs per (TM+TN) fragment loads
// (32 FLOP/B of L2 traffic at 4x4).  act: 0 none, 1 silu, 2 softplus.
// __launch_bounds__(128,1): 4 waves/block, min 1 block/CU, so the
// compiler may allocate a large per-wave VGPR file (no accumulator spills).
// =====================================================================
template<int TM, int TN>
__global__ __launch_bounds__(128, 1)
void k_gemm_t(const _Float16* __restrict__ A, int lda, long long sA,
              const _Float16* __restrict__ Bt, int ldb, long long sB,
              const float* __restrict__ bias,
              float* __restrict__ Cf, long long sCf,
              _Float16* __restrict__ Ch, long long sCh,
              int ldc, int M, int N, int K, int act)
{
    const int batch = blockIdx.y;
    A  += (long long)batch * sA;
    Bt += (long long)batch * sB;

    const int tilesN = N / (16 * TN);
    const int tilesM = M / (16 * TM);
    const int wid = blockIdx.x * 4 + (threadIdx.x >> 5);
    if (wid >= tilesM * tilesN) return;          // wave-uniform
    const int tm = wid / tilesN;
    const int tn = wid % tilesN;

    const int lane = threadIdx.x & 31;
    const int l15  = lane & 15;
    const int kh   = lane >> 4;

    const _Float16* Ar[TM];
    const _Float16* Br[TN];
    #pragma unroll
    for (int im = 0; im < TM; ++im)
        Ar[im] = A + (size_t)((tm * TM + im) * 16 + l15) * lda;
    #pragma unroll
    for (int in = 0; in < TN; ++in)
        Br[in] = Bt + (size_t)((tn * TN + in) * 16 + l15) * ldb;

    v8f acc[TM][TN];
    #pragma unroll
    for (int im = 0; im < TM; ++im)
        #pragma unroll
        for (int in = 0; in < TN; ++in)
            acc[im][in] = v8f{};

    for (int k0 = 0; k0 < K; k0 += 32) {
        v16h af[TM], bf[TN];
        #pragma unroll
        for (int im = 0; im < TM; ++im) {
            // A frag 16x32: lanes 0-15 K 0..7 & 16..23, lanes 16-31 K 8..15 & 24..31
            v8h alo = *(const v8h*)(Ar[im] + k0 + kh * 8);
            v8h ahi = *(const v8h*)(Ar[im] + k0 + 16 + kh * 8);
            af[im] = __builtin_shufflevector(alo, ahi,
                          0,1,2,3,4,5,6,7,8,9,10,11,12,13,14,15);
        }
        #pragma unroll
        for (int in = 0; in < TN; ++in) {
            // B frag 32x16: lanes 0-15 K 0..7 & 16..23, lanes 16-31 K 8..15 & 24..31
            v8h blo = *(const v8h*)(Br[in] + k0 + kh * 8);
            v8h bhi = *(const v8h*)(Br[in] + k0 + 16 + kh * 8);
            bf[in] = __builtin_shufflevector(blo, bhi,
                          0,1,2,3,4,5,6,7,8,9,10,11,12,13,14,15);
        }
        #pragma unroll
        for (int im = 0; im < TM; ++im)
            #pragma unroll
            for (int in = 0; in < TN; ++in)
                acc[im][in] = __builtin_amdgcn_wmma_f32_16x16x32_f16(
                    false, af[im], false, bf[in], (short)0, acc[im][in],
                    false, false);
        __builtin_prefetch(Ar[0] + k0 + 128, 0, 0);
        __builtin_prefetch(Br[0] + k0 + 128, 0, 0);
    }

    #pragma unroll
    for (int in = 0; in < TN; ++in) {
        const int col = (tn * TN + in) * 16 + l15;
        const float bv = bias ? bias[col] : 0.f;
        #pragma unroll
        for (int im = 0; im < TM; ++im) {
            #pragma unroll
            for (int r = 0; r < 8; ++r) {
                const int row = (tm * TM + im) * 16 + r + 8 * kh;
                float v = acc[im][in][r] + bv;
                if (act == 1)      v = siluf(v);
                else if (act == 2) v = (v > 20.f) ? v : log1pf(__expf(v));
                if (Cf) Cf[(long long)batch * sCf + (size_t)row * ldc + col] = v;
                if (Ch) Ch[(long long)batch * sCh + (size_t)row * ldc + col] = (_Float16)v;
            }
        }
    }
}

// f32 -> f16 convert with optional zero-padding of columns
__global__ void k_cvt(const float* __restrict__ src, int srcStride,
                      _Float16* __restrict__ dst, int dstStride,
                      int cols, int n)
{
    int idx = blockIdx.x * blockDim.x + threadIdx.x;
    if (idx >= n) return;
    int r = idx / dstStride, c = idx % dstStride;
    dst[idx] = (c < cols) ? (_Float16)src[(size_t)r * srcStride + c] : (_Float16)0.f;
}

// Z-order permutations (8 x 256)
__global__ void k_perms(int* __restrict__ perms)
{
    int i = blockIdx.x, p = threadIdx.x;
    int r = p >> 4, c = p & 15;
    auto tmpv = [](int rr, int cc) { return rr * 16 + ((rr & 1) ? (15 - cc) : cc); };
    int v;
    switch (i & 3) {
        case 1:  v = tmpv(r, c);        break;
        case 2:  v = tmpv(c, r);        break;
        case 3:  v = tmpv(r, 15 - c);   break;
        default: v = tmpv(15 - c, r);   break;
    }
    perms[i * 256 + p] = v;
}

// Patch embed + 2D sincos positional embedding
__global__ void k_patch(const float* __restrict__ x, const float* __restrict__ pw,
                        const float* __restrict__ pb, float* __restrict__ h)
{
    int idx = blockIdx.x * blockDim.x + threadIdx.x;
    if (idx >= TOK * DD) return;
    int d = idx % DD;
    int s = (idx / DD) % SS;
    int b = idx / (DD * SS);
    int gi = s >> 4, gj = s & 15;
    float acc = pb[d];
    #pragma unroll
    for (int c = 0; c < CIN; ++c)
        #pragma unroll
        for (int p = 0; p < PP; ++p)
            #pragma unroll
            for (int q = 0; q < PP; ++q)
                acc += x[((b * CIN + c) * IMG + (gi * 2 + p)) * IMG + (gj * 2 + q)]
                     * pw[((d * CIN + c) * PP + p) * PP + q];
    float coord = (d < 384) ? (float)gj : (float)gi;
    int dd = d % 384;
    int k  = dd % 192;
    float omega = __expf(-9.2103404f * (float)k / 192.f);
    float arg = coord * omega;
    float pos = (dd < 192) ? __sinf(arg) : __cosf(arg);
    h[idx] = acc + pos;
}

// timestep frequency embedding [B, FREQ]
__global__ void k_tembin(const float* __restrict__ t, float* __restrict__ out)
{
    int idx = blockIdx.x * blockDim.x + threadIdx.x;
    if (idx >= BB * FREQ) return;
    int b = idx / FREQ, c = idx % FREQ;
    int k = c % 128;
    float freq = __expf(-9.2103404f * (float)k / 128.f);
    float arg = t[b] * freq;
    out[idx] = (c < 128) ? __cosf(arg) : __sinf(arg);
}

// c = concat(temb + y, y2);  csilu16 = silu(c) in f16
__global__ void k_cvec(const float* __restrict__ temb, const float* __restrict__ y,
                       const float* __restrict__ y2, _Float16* __restrict__ csilu16)
{
    int idx = blockIdx.x * blockDim.x + threadIdx.x;
    if (idx >= BB * DI) return;
    int b = idx / DI, d = idx % DI;
    float v = (d < DD) ? (temb[b * DD + d] + y[b * DD + d]) : y2[b * DD + (d - DD)];
    csilu16[idx] = (_Float16)siluf(v);
}

// LayerNorm(D) + adaLN modulate -> f16 (gamma/beta may be null => 1/0)
__global__ void k_lnmod(const float* __restrict__ x,
                        const float* __restrict__ gamma, const float* __restrict__ beta,
                        const float* __restrict__ mod, int modStride, int shOff, int scOff,
                        _Float16* __restrict__ out, float eps)
{
    int row = blockIdx.x;        // b*S + s
    int b = row >> 8;
    int tid = threadIdx.x;
    __shared__ float sm[256], sv[256];
    float s1 = 0.f, s2 = 0.f;
    for (int d = tid; d < DD; d += 256) {
        float v = x[(size_t)row * DD + d];
        s1 += v; s2 += v * v;
    }
    sm[tid] = s1; sv[tid] = s2; __syncthreads();
    for (int o = 128; o > 0; o >>= 1) {
        if (tid < o) { sm[tid] += sm[tid + o]; sv[tid] += sv[tid + o]; }
        __syncthreads();
    }
    float mean = sm[0] / DD;
    float var  = sv[0] / DD - mean * mean;
    float rs   = rsqrtf(var + eps);
    for (int d = tid; d < DD; d += 256) {
        float v = (x[(size_t)row * DD + d] - mean) * rs;
        if (gamma) v = v * gamma[d] + beta[d];
        float scv = mod[(size_t)b * modStride + scOff + d];
        float shv = mod[(size_t)b * modStride + shOff + d];
        out[(size_t)row * DD + d] = (_Float16)(v * (1.f + scv) + shv);
    }
}

// split qkv [B,S,3,H,HD] -> q16 (scaled), k16 [B,H,S,HD], vT16 [B,H,HD,S]
__global__ void k_split(const float* __restrict__ qkv, _Float16* __restrict__ q16,
                        _Float16* __restrict__ k16, _Float16* __restrict__ vT16)
{
    int idx = blockIdx.x * blockDim.x + threadIdx.x;
    if (idx >= TOK * 3 * DD) return;
    int hd = idx % HD;
    int hh = (idx / HD) % HEADS;
    int which = (idx / DD) % 3;
    int s = (idx / (3 * DD)) % SS;
    int b = idx / (3 * DD * SS);
    float v = qkv[idx];
    size_t bh = (size_t)b * HEADS + hh;
    if (which == 0)      q16[(bh * SS + s) * HD + hd] = (_Float16)(v * 0.1020620726f);
    else if (which == 1) k16[(bh * SS + s) * HD + hd] = (_Float16)v;
    else                 vT16[(bh * HD + hd) * SS + s] = (_Float16)v;
}

// softmax over last dim (S=256), f32 in -> f16 out
__global__ void k_softmax(const float* __restrict__ att, _Float16* __restrict__ att16)
{
    int row = blockIdx.x;
    int tid = threadIdx.x;
    __shared__ float red[256];
    float v = att[(size_t)row * SS + tid];
    red[tid] = v; __syncthreads();
    for (int o = 128; o > 0; o >>= 1) {
        if (tid < o) red[tid] = fmaxf(red[tid], red[tid + o]);
        __syncthreads();
    }
    float mx = red[0]; __syncthreads();
    float e = __expf(v - mx);
    red[tid] = e; __syncthreads();
    for (int o = 128; o > 0; o >>= 1) {
        if (tid < o) red[tid] += red[tid + o];
        __syncthreads();
    }
    att16[(size_t)row * SS + tid] = (_Float16)(e / red[0]);
}

// [B,H,S,HD] f16 -> [B,S,H*HD] f16
__global__ void k_oresh(const _Float16* __restrict__ o16, _Float16* __restrict__ ar16)
{
    int idx = blockIdx.x * blockDim.x + threadIdx.x;
    if (idx >= TOK * DD) return;
    int hd = idx % HD;
    int s  = (idx / HD) % SS;
    int hh = (idx / (HD * SS)) % HEADS;
    int b  = idx / (HD * SS * HEADS);
    ar16[((size_t)b * SS + s) * DD + hh * HD + hd] = o16[idx];
}

// xp16[b,s,:] = xn16[b,perm[s],:]
__global__ void k_permgather(const _Float16* __restrict__ xn16, const int* __restrict__ perm,
                             _Float16* __restrict__ xp16)
{
    int idx = blockIdx.x * blockDim.x + threadIdx.x;
    if (idx >= TOK * DD) return;
    int d = idx % DD;
    int s = (idx / DD) % SS;
    int b = idx / (DD * SS);
    xp16[idx] = xn16[((size_t)b * SS + perm[s]) * DD + d];
}

// causal depthwise conv over sequence + silu; writes f32 and f16
__global__ void k_conv(const float* __restrict__ xz, const float* __restrict__ cw,
                       const float* __restrict__ cb,
                       float* __restrict__ xc, _Float16* __restrict__ xc16)
{
    int idx = blockIdx.x * blockDim.x + threadIdx.x;
    if (idx >= TOK * DI) return;
    int ch = idx % DI;
    int s  = (idx / DI) % SS;
    int b  = idx / (DI * SS);
    float acc = cb[ch];
    #pragma unroll
    for (int k = 0; k < DCONV; ++k) {
        int sp = s - (DCONV - 1) + k;
        if (sp >= 0)
            acc += xz[((size_t)b * SS + sp) * (2 * DI) + ch] * cw[ch * DCONV + k];
    }
    float v = siluf(acc);
    xc[idx]   = v;
    xc16[idx] = (_Float16)v;
}

// selective-scan: one thread per (b, ch), 16-wide state in registers
__global__ void k_scan(const float* __restrict__ dt, const float* __restrict__ dbl,
                       const float* __restrict__ xc, const float* __restrict__ xz,
                       const float* __restrict__ A_log, const float* __restrict__ Dp,
                       _Float16* __restrict__ y16)
{
    int idx = blockIdx.x * blockDim.x + threadIdx.x;
    if (idx >= BB * DI) return;
    int ch = idx % DI;
    int b  = idx / DI;
    float Af[DS2], h[DS2];
    #pragma unroll
    for (int j = 0; j < DS2; ++j) { Af[j] = -__expf(A_log[ch * DS2 + j]); h[j] = 0.f; }
    float dpv = Dp[ch];
    for (int s = 0; s < SS; ++s) {
        size_t r = (size_t)b * SS + s;
        float dtv = dt[r * DI + ch];
        float xv  = xc[r * DI + ch];
        const float* dr = dbl + r * (DTR + 2 * DS2);
        float acc = 0.f;
        #pragma unroll
        for (int j = 0; j < DS2; ++j) {
            h[j] = __expf(dtv * Af[j]) * h[j] + dtv * dr[DTR + j] * xv;
            acc += h[j] * dr[DTR + DS2 + j];
        }
        float zm = xz[r * (2 * DI) + DI + ch];
        y16[r * DI + ch] = (_Float16)((acc + dpv * xv) * siluf(zm));
    }
}

// build comb = [m (perm-gathered), a], LN(2D) with fn_g/fn_b -> f16
__global__ void k_combln(const float* __restrict__ mo, const float* __restrict__ ao,
                         const int* __restrict__ perm,
                         const float* __restrict__ g2, const float* __restrict__ b2,
                         _Float16* __restrict__ out)
{
    int row = blockIdx.x;
    int b = row >> 8, s = row & 255;
    int tid = threadIdx.x;
    int sp = perm[s];
    __shared__ float sm[256], sv[256];
    float vals[6];
    float s1 = 0.f, s2 = 0.f;
    #pragma unroll
    for (int i = 0; i < 6; ++i) {
        int d = tid + i * 256;
        float v = (d < DD) ? mo[((size_t)b * SS + sp) * DD + d]
                           : ao[(size_t)row * DD + (d - DD)];
        vals[i] = v; s1 += v; s2 += v * v;
    }
    sm[tid] = s1; sv[tid] = s2; __syncthreads();
    for (int o = 128; o > 0; o >>= 1) {
        if (tid < o) { sm[tid] += sm[tid + o]; sv[tid] += sv[tid + o]; }
        __syncthreads();
    }
    float mean = sm[0] / DI;
    float var  = sv[0] / DI - mean * mean;
    float rs   = rsqrtf(var + 1e-5f);
    #pragma unroll
    for (int i = 0; i < 6; ++i) {
        int d = tid + i * 256;
        out[(size_t)row * DI + d] = (_Float16)((vals[i] - mean) * rs * g2[d] + b2[d]);
    }
}

// h += gate[b,d] * fnout (fnout already silu'd by GEMM epilogue)
__global__ void k_resid(float* __restrict__ h, const float* __restrict__ adaout,
                        const float* __restrict__ fnout)
{
    int idx = blockIdx.x * blockDim.x + threadIdx.x;
    if (idx >= TOK * DD) return;
    int d = idx % DD;
    int b = idx / (DD * SS);
    h[idx] += adaout[(size_t)b * (3 * DD) + 2 * DD + d] * fnout[idx];
}

// unpatchify: fo [B,S,32] -> [B,OUTC,32,32]
__global__ void k_unpatch(const float* __restrict__ fo, float* __restrict__ out)
{
    int idx = blockIdx.x * blockDim.x + threadIdx.x;
    if (idx >= BB * OUTC * IMG * IMG) return;
    int ww = idx % IMG;
    int hh = (idx / IMG) % IMG;
    int c  = (idx / (IMG * IMG)) % OUTC;
    int b  = idx / (IMG * IMG * OUTC);
    int q = ww & 1, j = ww >> 1;
    int p = hh & 1, i = hh >> 1;
    out[idx] = fo[((size_t)b * SS + (i * NN + j)) * (PP * PP * OUTC) + (p * PP + q) * OUTC + c];
}

// =====================================================================
// Host orchestration
// =====================================================================
static inline void gemm(hipStream_t st,
                        const _Float16* A, int lda, long long sA,
                        const _Float16* Bt, int ldb, long long sB,
                        const float* bias,
                        float* Cf, long long sCf, _Float16* Ch, long long sCh,
                        int ldc, int M, int N, int K, int act, int batch)
{
    if ((M % 64 == 0) && (N % 64 == 0)) {
        int tiles = (M / 64) * (N / 64);
        dim3 g((tiles + 3) / 4, batch), b(128);
        k_gemm_t<4, 4><<<g, b, 0, st>>>(A, lda, sA, Bt, ldb, sB, bias,
                                        Cf, sCf, Ch, sCh, ldc, M, N, K, act);
    } else if ((M % 32 == 0) && (N % 32 == 0)) {
        int tiles = (M / 32) * (N / 32);
        dim3 g((tiles + 3) / 4, batch), b(128);
        k_gemm_t<2, 2><<<g, b, 0, st>>>(A, lda, sA, Bt, ldb, sB, bias,
                                        Cf, sCf, Ch, sCh, ldc, M, N, K, act);
    } else {
        int tiles = (M / 16) * (N / 16);
        dim3 g((tiles + 3) / 4, batch), b(128);
        k_gemm_t<1, 1><<<g, b, 0, st>>>(A, lda, sA, Bt, ldb, sB, bias,
                                        Cf, sCf, Ch, sCh, ldc, M, N, K, act);
    }
}

static inline void cvt(hipStream_t st, const float* src, int srcStride,
                       _Float16* dst, int dstStride, int cols, int rows)
{
    int n = rows * dstStride;
    k_cvt<<<(n + 255) / 256, 256, 0, st>>>(src, srcStride, dst, dstStride, cols, n);
}

extern "C" void kernel_launch(void* const* d_in, const int* in_sizes, int n_in,
                              void* d_out, int out_size, void* d_ws, size_t ws_size,
                              hipStream_t stream)
{
    (void)in_sizes; (void)n_in; (void)out_size; (void)ws_size;

    const float* in_x  = (const float*)d_in[0];
    const float* in_t  = (const float*)d_in[1];
    const float* in_y  = (const float*)d_in[2];
    const float* in_y2 = (const float*)d_in[3];
    const float* patch_w = (const float*)d_in[4];
    const float* patch_b = (const float*)d_in[5];
    const float* t_w1 = (const float*)d_in[6];
    const float* t_b1 = (const float*)d_in[7];
    const float* t_w2 = (const float*)d_in[8];
    const float* t_b2 = (const float*)d_in[9];
    auto BP = [&](int i, int j) { return (const float*)d_in[10 + i * 21 + j]; };
    const float* fin_ada_w = (const float*)d_in[10 + DEPTH * 21 + 0];
    const float* fin_ada_b = (const float*)d_in[10 + DEPTH * 21 + 1];
    const float* fin_w     = (const float*)d_in[10 + DEPTH * 21 + 2];
    const float* fin_b     = (const float*)d_in[10 + DEPTH * 21 + 3];

    char* ws = (char*)d_ws;
    size_t off = 0;
    auto alloc = [&](size_t bytes) -> void* {
        void* p = ws + off;
        off += (bytes + 255) & ~(size_t)255;
        return p;
    };

    int*        perms  = (int*)      alloc(DEPTH * SS * 4);
    float*      h      = (float*)    alloc((size_t)TOK * DD * 4);
    float*      tin    = (float*)    alloc(BB * FREQ * 4);
    _Float16*   tin16  = (_Float16*) alloc(BB * FREQ * 2);
    _Float16*   t116   = (_Float16*) alloc(BB * DD * 2);
    float*      temb   = (float*)    alloc(BB * DD * 4);
    _Float16*   csilu16= (_Float16*) alloc(BB * DI * 2);
    float*      adaout = (float*)    alloc(BB * 3 * DD * 4);
    float*      finada = (float*)    alloc(BB * DI * 4);
    _Float16*   xn16   = (_Float16*) alloc((size_t)TOK * DD * 2);
    float*      qkv    = (float*)    alloc((size_t)TOK * 3 * DD * 4);
    _Float16*   q16    = (_Float16*) alloc((size_t)TOK * DD * 2);
    _Float16*   k16    = (_Float16*) alloc((size_t)TOK * DD * 2);
    _Float16*   vT16   = (_Float16*) alloc((size_t)TOK * DD * 2);
    float*      att    = (float*)    alloc((size_t)BB * HEADS * SS * SS * 4);
    _Float16*   att16  = (_Float16*) alloc((size_t)BB * HEADS * SS * SS * 2);
    _Float16*   o16    = (_Float16*) alloc((size_t)TOK * DD * 2);
    _Float16*   ar16   = (_Float16*) alloc((size_t)TOK * DD * 2);
    float*      aout   = (float*)    alloc((size_t)TOK * DD * 4);
    _Float16*   xp16   = (_Float16*) alloc((size_t)TOK * DD * 2);
    float*      xz     = (float*)    alloc((size_t)TOK * 2 * DI * 4);
    float*      xc     = (float*)    alloc((size_t)TOK * DI * 4);
    _Float16*   xc16   = (_Float16*) alloc((size_t)TOK * DI * 2);
    float*      dbl    = (float*)    alloc((size_t)TOK * (DTR + 2 * DS2) * 4);
    _Float16*   dtin16 = (_Float16*) alloc((size_t)TOK * 64 * 2);
    _Float16*   dtw16  = (_Float16*) alloc((size_t)DI * 64 * 2);
    float*      dtbuf  = (float*)    alloc((size_t)TOK * DI * 4);
    _Float16*   y16    = (_Float16*) alloc((size_t)TOK * DI * 2);
    float*      mo     = (float*)    alloc((size_t)TOK * DD * 4);
    _Float16*   comb16 = (_Float16*) alloc((size_t)TOK * DI * 2);
    float*      fnout  = (float*)    alloc((size_t)TOK * DD * 4);
    _Float16*   hf16   = (_Float16*) alloc((size_t)TOK * DD * 2);
    float*      fo     = (float*)    alloc((size_t)TOK * PP * PP * OUTC * 4);
    _Float16*   wbuf   = (_Float16*) alloc((size_t)3 * DD * DI * 2);   // max weight

    // ---- constants / embeddings ----
    k_perms<<<DEPTH, 256, 0, stream>>>(perms);
    {
        int n = TOK * DD;
        k_patch<<<(n + 255) / 256, 256, 0, stream>>>(in_x, patch_w, patch_b, h);
    }
    k_tembin<<<(BB * FREQ + 255) / 256, 256, 0, stream>>>(in_t, tin);
    cvt(stream, tin, FREQ, tin16, FREQ, FREQ, BB);
    cvt(stream, t_w1, FREQ, wbuf, FREQ, FREQ, DD);
    gemm(stream, tin16, FREQ, 0, wbuf, FREQ, 0, t_b1,
         nullptr, 0, t116, 0, DD, BB, DD, FREQ, /*silu*/1, 1);
    cvt(stream, t_w2, DD, wbuf, DD, DD, DD);
    gemm(stream, t116, DD, 0, wbuf, DD, 0, t_b2,
         temb, 0, nullptr, 0, DD, BB, DD, DD, 0, 1);
    k_cvec<<<(BB * DI + 255) / 256, 256, 0, stream>>>(temb, in_y, in_y2, csilu16);

    const int nTokD  = TOK * DD;
    const int nTokDI = TOK * DI;

    // ---- transformer/mamba blocks ----
    for (int i = 0; i < DEPTH; ++i) {
        const int* perm = perms + i * SS;

        // adaLN: silu(c) @ ada_w^T + ada_b  -> [B, 3D]
        cvt(stream, BP(i, 6), DI, wbuf, DI, DI, 3 * DD);
        gemm(stream, csilu16, DI, 0, wbuf, DI, 0, BP(i, 7),
             adaout, 0, nullptr, 0, 3 * DD, BB, 3 * DD, DI, 0, 1);

        // xn = modulate(LN(h))
        k_lnmod<<<TOK, 256, 0, stream>>>(h, BP(i, 0), BP(i, 1),
                                         adaout, 3 * DD, 0, DD, xn16, 1e-5f);

        // ---- attention ----
        cvt(stream, BP(i, 2), DD, wbuf, DD, DD, 3 * DD);
        gemm(stream, xn16, DD, 0, wbuf, DD, 0, BP(i, 3),
             qkv, 0, nullptr, 0, 3 * DD, TOK, 3 * DD, DD, 0, 1);
        {
            int n = TOK * 3 * DD;
            k_split<<<(n + 255) / 256, 256, 0, stream>>>(qkv, q16, k16, vT16);
        }
        // scores: batched [S,HD] x [S,HD]^T
        gemm(stream, q16, HD, (long long)SS * HD, k16, HD, (long long)SS * HD, nullptr,
             att, (long long)SS * SS, nullptr, 0, SS, SS, SS, HD, 0, BB * HEADS);
        k_softmax<<<BB * HEADS * SS, 256, 0, stream>>>(att, att16);
        // o = att @ v : batched [S,S] x [HD,S]^T
        gemm(stream, att16, SS, (long long)SS * SS, vT16, SS, (long long)HD * SS, nullptr,
             nullptr, 0, o16, (long long)SS * HD, HD, SS, HD, SS, 0, BB * HEADS);
        k_oresh<<<(nTokD + 255) / 256, 256, 0, stream>>>(o16, ar16);
        cvt(stream, BP(i, 4), DD, wbuf, DD, DD, DD);
        gemm(stream, ar16, DD, 0, wbuf, DD, 0, BP(i, 5),
             aout, 0, nullptr, 0, DD, TOK, DD, DD, 0, 1);

        // ---- mamba ----
        k_permgather<<<(nTokD + 255) / 256, 256, 0, stream>>>(xn16, perm, xp16);
        cvt(stream, BP(i, 8), DD, wbuf, DD, DD, 2 * DI);
        gemm(stream, xp16, DD, 0, wbuf, DD, 0, nullptr,
             xz, 0, nullptr, 0, 2 * DI, TOK, 2 * DI, DD, 0, 1);
        k_conv<<<(nTokDI + 255) / 256, 256, 0, stream>>>(xz, BP(i, 9), BP(i, 10), xc, xc16);
        cvt(stream, BP(i, 11), DI, wbuf, DI, DI, DTR + 2 * DS2);
        gemm(stream, xc16, DI, 0, wbuf, DI, 0, nullptr,
             dbl, 0, nullptr, 0, DTR + 2 * DS2, TOK, DTR + 2 * DS2, DI, 0, 1);
        // dt = softplus(dbl[:, :48] @ dt_w^T + dt_b)   (K padded 48 -> 64)
        cvt(stream, dbl, DTR + 2 * DS2, dtin16, 64, DTR, TOK);
        cvt(stream, BP(i, 12), DTR, dtw16, 64, DTR, DI);
        gemm(stream, dtin16, 64, 0, dtw16, 64, 0, BP(i, 13),
             dtbuf, 0, nullptr, 0, DI, TOK, DI, 64, /*softplus*/2, 1);
        k_scan<<<(BB * DI + 255) / 256, 256, 0, stream>>>(dtbuf, dbl, xc, xz,
                                                          BP(i, 14), BP(i, 15), y16);
        cvt(stream, BP(i, 16), DI, wbuf, DI, DI, DD);
        gemm(stream, y16, DI, 0, wbuf, DI, 0, nullptr,
             mo, 0, nullptr, 0, DD, TOK, DD, DI, 0, 1);

        // ---- fuse + MLP + residual ----
        k_combln<<<TOK, 256, 0, stream>>>(mo, aout, perm, BP(i, 17), BP(i, 18), comb16);
        cvt(stream, BP(i, 19), DI, wbuf, DI, DI, DD);
        gemm(stream, comb16, DI, 0, wbuf, DI, 0, BP(i, 20),
             fnout, 0, nullptr, 0, DD, TOK, DD, DI, /*silu*/1, 1);
        k_resid<<<(nTokD + 255) / 256, 256, 0, stream>>>(h, adaout, fnout);
    }

    // ---- final layer ----
    cvt(stream, fin_ada_w, DI, wbuf, DI, DI, DI);
    gemm(stream, csilu16, DI, 0, wbuf, DI, 0, fin_ada_b,
         finada, 0, nullptr, 0, DI, BB, DI, DI, 0, 1);
    k_lnmod<<<TOK, 256, 0, stream>>>(h, nullptr, nullptr,
                                     finada, DI, 0, DD, hf16, 1e-6f);
    cvt(stream, fin_w, DD, wbuf, DD, DD, PP * PP * OUTC);
    gemm(stream, hf16, DD, 0, wbuf, DD, 0, fin_b,
         fo, 0, nullptr, 0, PP * PP * OUTC, TOK, PP * PP * OUTC, DD, 0, 1);
    {
        int n = BB * OUTC * IMG * IMG;
        k_unpatch<<<(n + 255) / 256, 256, 0, stream>>>(fo, (float*)d_out);
    }
}
